// GatedTransformerXLLayer_29772713295988
// MI455X (gfx1250) — matrix operand order repaired and checked by reading, not compiled, not run on hardware
//
#include <hip/hip_runtime.h>
#include <hip/hip_bf16.h>

// ---------------------------------------------------------------------------
// Types for CDNA5 WMMA (wave32, 16x16x32 bf16 -> f32)
// ---------------------------------------------------------------------------
typedef __bf16 bf16_t;
typedef __attribute__((ext_vector_type(16))) __bf16 v16bf;
typedef __attribute__((ext_vector_type(8)))  float  v8f;

union AFrag { v16bf v; unsigned u[8]; };
union CFrag { v8f v; float f[8]; };

// Tensor Data Mover support (probe-verified builtins; this toolchain = 6-arg form)
#if defined(__has_builtin)
# if __has_builtin(__builtin_amdgcn_tensor_load_to_lds) && __has_builtin(__builtin_amdgcn_s_wait_tensorcnt)
#  define USE_TDM 1
# endif
#endif

#ifdef USE_TDM
typedef unsigned int u32x4 __attribute__((ext_vector_type(4)));
typedef int          i32x4 __attribute__((ext_vector_type(4)));
typedef int          i32x8 __attribute__((ext_vector_type(8)));
#endif

__device__ __forceinline__ bf16_t f2bf(float f) {
  unsigned u; __builtin_memcpy(&u, &f, 4);
  unsigned r = u + 0x7FFFu + ((u >> 16) & 1u);   // round-to-nearest-even
  unsigned short hs = (unsigned short)(r >> 16);
  bf16_t o; __builtin_memcpy(&o, &hs, 2); return o;
}
__device__ __forceinline__ float bf2f(bf16_t b) {
  unsigned short hs; __builtin_memcpy(&hs, &b, 2);
  unsigned u = ((unsigned)hs) << 16; float f; __builtin_memcpy(&f, &u, 4); return f;
}

// ---------------------------------------------------------------------------
// Generic bf16 WMMA GEMM: C[M,N] = act( A[M,K] @ W[K,N] + bias + Cin )
// W is supplied PRE-TRANSPOSED as WT[N,K] so both tiles stage as b128 copies.
// Block tile 128x128x32, 256 threads = 8 waves (4 row x 2 col strips),
// each wave owns a 32x64 sub-tile = 8 wmma accumulators.
// Double-buffered LDS; next k-slice is loaded into registers before the
// barrier so global latency overlaps the WMMA stream.
// ---------------------------------------------------------------------------
#define GBM 128
#define GBN 128
#define GBK 32

__global__ __launch_bounds__(256) void gemm_bf16_kernel(
    const bf16_t* __restrict__ A, const bf16_t* __restrict__ WT,
    const float* __restrict__ bias, const float* Cin,
    float* Cout, int M, int N, int K, int act)
{
  __shared__ bf16_t sA[2][GBM][GBK];   // [m][k]
  __shared__ bf16_t sB[2][GBN][GBK];   // [n][k]

  const int tid = threadIdx.x;
  const int wave = tid >> 5, lane = tid & 31;
  const int lane16 = lane & 15, laneHi = lane >> 4;
  const int wm = wave & 3, wn = wave >> 2;
  const int bm0 = blockIdx.y * GBM, bn0 = blockIdx.x * GBN;

  CFrag acc[2][4];
#pragma unroll
  for (int ms = 0; ms < 2; ++ms)
#pragma unroll
    for (int ns = 0; ns < 4; ++ns)
#pragma unroll
      for (int e = 0; e < 8; ++e) acc[ms][ns].f[e] = 0.0f;

  const int s_row = tid >> 1, s_k0 = (tid & 1) * 16;   // 128 rows x 32 k (16 elems/thread)
  const bf16_t* gA = A  + (size_t)(bm0 + s_row) * K + s_k0;
  const bf16_t* gB = WT + (size_t)(bn0 + s_row) * K + s_k0;

  uint4 ra0 = ((const uint4*)gA)[0], ra1 = ((const uint4*)(gA + 8))[0];
  uint4 rb0 = ((const uint4*)gB)[0], rb1 = ((const uint4*)(gB + 8))[0];

  const int nk = K / GBK;
  for (int kt = 0; kt < nk; ++kt) {
    const int buf = kt & 1;
    *(uint4*)&sA[buf][s_row][s_k0]     = ra0;
    *(uint4*)&sA[buf][s_row][s_k0 + 8] = ra1;
    *(uint4*)&sB[buf][s_row][s_k0]     = rb0;
    *(uint4*)&sB[buf][s_row][s_k0 + 8] = rb1;
    if (kt + 1 < nk) {
      const bf16_t* nA = gA + (size_t)(kt + 1) * GBK;
      const bf16_t* nB = gB + (size_t)(kt + 1) * GBK;
      ra0 = ((const uint4*)nA)[0]; ra1 = ((const uint4*)(nA + 8))[0];
      rb0 = ((const uint4*)nB)[0]; rb1 = ((const uint4*)(nB + 8))[0];
    }
    __syncthreads();

    AFrag afr[2], bfr[4];
#pragma unroll
    for (int ms = 0; ms < 2; ++ms) {
      const int row = wm * 32 + ms * 16 + lane16;
#pragma unroll
      for (int p = 0; p < 8; ++p) {
        // ISA A-layout (16-bit 16x32): lane<16 holds K {0..7,16..23}, lane>=16 +8
        const int kk = ((p < 4) ? (2 * p) : (16 + 2 * (p - 4))) + laneHi * 8;
        afr[ms].u[p] = *(const unsigned*)&sA[buf][row][kk];
      }
    }
#pragma unroll
    for (int ns = 0; ns < 4; ++ns) {
      const int col = wn * 64 + ns * 16 + lane16;
#pragma unroll
      for (int p = 0; p < 8; ++p) {
        // B-layout (32x16): same K order as A: lane<16 K {0..7,16..23}, lane>=16 +8
        const int kk = ((p < 4) ? (2 * p) : (16 + 2 * (p - 4))) + laneHi * 8;
        bfr[ns].u[p] = *(const unsigned*)&sB[buf][col][kk];
      }
    }
#pragma unroll
    for (int ms = 0; ms < 2; ++ms)
#pragma unroll
      for (int ns = 0; ns < 4; ++ns)
        acc[ms][ns].v = __builtin_amdgcn_wmma_f32_16x16x32_bf16(
            false, afr[ms].v, false, bfr[ns].v, (short)0, acc[ms][ns].v, false, false);
  }

  // epilogue: C layout = lane16 -> col, (laneHi*8 + e) -> row
#pragma unroll
  for (int ms = 0; ms < 2; ++ms)
#pragma unroll
    for (int ns = 0; ns < 4; ++ns) {
      const int col = bn0 + wn * 64 + ns * 16 + lane16;
#pragma unroll
      for (int e = 0; e < 8; ++e) {
        const int row = bm0 + wm * 32 + ms * 16 + laneHi * 8 + e;
        float v = acc[ms][ns].f[e];
        if (bias) v += bias[col];
        if (Cin)  v += Cin[(size_t)row * N + col];
        if (act == 1) v = fmaxf(v, 0.0f);
        Cout[(size_t)row * N + col] = v;
      }
    }
}

// ---------------------------------------------------------------------------
// fp32 [K,N] -> bf16 transposed [N,K], LDS-tiled so both sides are coalesced.
// ---------------------------------------------------------------------------
__global__ __launch_bounds__(256) void tconv_kernel(
    const float* __restrict__ src, bf16_t* __restrict__ dst, int K, int N)
{
  __shared__ bf16_t tile[32][33];
  const int bx = blockIdx.x * 32;  // n base
  const int by = blockIdx.y * 32;  // k base
  const int tx = threadIdx.x & 31, ty = threadIdx.x >> 5;
#pragma unroll
  for (int r = 0; r < 32; r += 8)
    tile[ty + r][tx] = f2bf(src[(size_t)(by + ty + r) * N + (bx + tx)]);
  __syncthreads();
#pragma unroll
  for (int r = 0; r < 32; r += 8)
    dst[(size_t)(bx + ty + r) * K + (by + tx)] = tile[tx][ty + r];
}

// ---------------------------------------------------------------------------
// Row LayerNorm over D=1024, output bf16. Row r < rowsA comes from srcA,
// otherwise from srcB (used for concat([memory, inputs])).
// ---------------------------------------------------------------------------
__global__ __launch_bounds__(256) void ln_rows_kernel(
    const float* __restrict__ srcA, const float* __restrict__ srcB, int rowsA,
    const float* __restrict__ g, const float* __restrict__ beta,
    bf16_t* __restrict__ out)
{
  __shared__ float red[256];
  const int row = blockIdx.x;
  const int t = threadIdx.x;
  const float* src = (row < rowsA) ? (srcA + (size_t)row * 1024)
                                   : (srcB + (size_t)(row - rowsA) * 1024);
  float x[4]; float s = 0.f;
#pragma unroll
  for (int i = 0; i < 4; ++i) { x[i] = src[t + 256 * i]; s += x[i]; }
  red[t] = s; __syncthreads();
  for (int o = 128; o > 0; o >>= 1) { if (t < o) red[t] += red[t + o]; __syncthreads(); }
  const float mu = red[0] * (1.0f / 1024.0f);
  __syncthreads();
  float s2 = 0.f;
#pragma unroll
  for (int i = 0; i < 4; ++i) { const float d = x[i] - mu; s2 += d * d; }
  red[t] = s2; __syncthreads();
  for (int o = 128; o > 0; o >>= 1) { if (t < o) red[t] += red[t + o]; __syncthreads(); }
  const float inv = rsqrtf(red[0] * (1.0f / 1024.0f) + 1e-6f);
#pragma unroll
  for (int i = 0; i < 4; ++i) {
    const int c = t + 256 * i;
    out[(size_t)row * 1024 + c] = f2bf((x[i] - mu) * inv * g[c] + beta[c]);
  }
}

// ---------------------------------------------------------------------------
// Elementwise / scatter kernels
// ---------------------------------------------------------------------------
__global__ void convert_f32_bf16_kernel(const float* __restrict__ s, bf16_t* __restrict__ d, int n) {
  for (int i = blockIdx.x * blockDim.x + threadIdx.x; i < n; i += blockDim.x * gridDim.x)
    d[i] = f2bf(s[i]);
}

// kv f32 [F*B, 2048] -> key bf16 [B][H][F][DH], value bf16 TRANSPOSED [B][H][DH][F]
__global__ void kv_scatter_kernel(const float* __restrict__ kv,
                                  bf16_t* __restrict__ keyb, bf16_t* __restrict__ valb) {
  const int n = 8 * 16 * 1024 * 64;
  for (int i = blockIdx.x * blockDim.x + threadIdx.x; i < n; i += blockDim.x * gridDim.x) {
    const int d = i & 63, h = (i >> 6) & 15, b = (i >> 10) & 7, f = i >> 13;
    const size_t srow = (size_t)(f * 8 + b) * 2048;
    const size_t bh = (size_t)b * 16 + h;
    keyb[(bh * 1024 + f) * 64 + d] = f2bf(kv[srow + h * 64 + d]);
    valb[(bh * 64 + d) * 1024 + f] = f2bf(kv[srow + 1024 + h * 64 + d]);
  }
}

// q f32 [Q*B, 1024] + u/v -> qu/qv bf16 [B][H][Q][DH]
__global__ void quqv_kernel(const float* __restrict__ qf, const float* __restrict__ u,
                            const float* __restrict__ v,
                            bf16_t* __restrict__ qub, bf16_t* __restrict__ qvb) {
  const int n = 512 * 8 * 16 * 64;
  for (int i = blockIdx.x * blockDim.x + threadIdx.x; i < n; i += blockDim.x * gridDim.x) {
    const int d = i & 63, h = (i >> 6) & 15, b = (i >> 10) & 7, q = i >> 13;
    const float q0 = qf[(size_t)(q * 8 + b) * 1024 + h * 64 + d];
    const size_t dst = ((((size_t)b * 16 + h) * 512) + q) * 64 + d;
    qub[dst] = f2bf(q0 + u[h * 64 + d]);
    qvb[dst] = f2bf(q0 + v[h * 64 + d]);
  }
}

// r f32 [F, 1024] -> rb bf16 [H][F][DH]
__global__ void r_scatter_kernel(const float* __restrict__ rf, bf16_t* __restrict__ rb) {
  const int n = 16 * 1024 * 64;
  for (int i = blockIdx.x * blockDim.x + threadIdx.x; i < n; i += blockDim.x * gridDim.x) {
    const int d = i & 63, f = (i >> 6) & 1023, h = i >> 16;
    rb[(((size_t)h * 1024) + f) * 64 + d] = f2bf(rf[(size_t)f * 1024 + h * 64 + d]);
  }
}

__global__ void sigmoid_kernel(float* t, const float* __restrict__ bsub, int n) {
  for (int i = blockIdx.x * blockDim.x + threadIdx.x; i < n; i += blockDim.x * gridDim.x) {
    float x = t[i];
    if (bsub) x -= bsub[i & 1023];
    t[i] = 1.0f / (1.0f + __expf(-x));
  }
}

__global__ void tanh_kernel(float* t, int n) {
  for (int i = blockIdx.x * blockDim.x + threadIdx.x; i < n; i += blockDim.x * gridDim.x)
    t[i] = tanhf(t[i]);
}

__global__ void rx_build_kernel(const float* __restrict__ rg, const float* __restrict__ x,
                                bf16_t* __restrict__ rxb, int n) {
  for (int i = blockIdx.x * blockDim.x + threadIdx.x; i < n; i += blockDim.x * gridDim.x)
    rxb[i] = f2bf(rg[i] * x[i]);
}

__global__ void gate_combine_kernel(const float* __restrict__ x, const float* __restrict__ z,
                                    const float* __restrict__ hh, float* __restrict__ out, int n) {
  for (int i = blockIdx.x * blockDim.x + threadIdx.x; i < n; i += blockDim.x * gridDim.x) {
    const float zi = z[i];
    out[i] = (1.0f - zi) * x[i] + zi * hh[i];
  }
}

// ---------------------------------------------------------------------------
// Fused Transformer-XL flash attention.
// Block = (b, h, 32-row q-tile), 256 threads = 8 waves (2 row x 4 col strips).
// Online softmax over 64-wide k-tiles. rel_shift folded in as
// pos[i,j] = band[i][ j + 31 - i ], band computed by WMMA against r rows
// [f0 + 511 - i0 - 31 .. +95] (clamped; overflow lands on masked entries).
// The 64x64 K-tile is staged by the Tensor Data Mover when available.
// ---------------------------------------------------------------------------
__global__ __launch_bounds__(256) void attn_kernel(
    const bf16_t* __restrict__ qub,   // [B][H][Q][DH]
    const bf16_t* __restrict__ qvb,   // [B][H][Q][DH]
    const bf16_t* __restrict__ keyb,  // [B][H][F][DH]
    const bf16_t* __restrict__ valT,  // [B][H][DH][F]  (pre-transposed)
    const bf16_t* __restrict__ rb,    // [H][F][DH]
    float* __restrict__ av)           // [Q*B][H*DH]
{
  __shared__ bf16_t s_qu[32][64];
  __shared__ bf16_t s_qv[32][64];
  __shared__ bf16_t s_k[64][64];     // B-layout [n=j][k=d]
  __shared__ bf16_t s_v[64][64];     // B-layout [n=d][k=j]
  __shared__ bf16_t s_r[96][64];     // B-layout [n=jj][k=d]
  __shared__ bf16_t s_band[32][104]; // position band (padded)
  __shared__ float  s_S[32][64];     // masked scaled scores
  __shared__ bf16_t s_p[32][64];     // Pexp, A-layout
  __shared__ float  s_m[32], s_l[32], s_c[32];

  const int tid = threadIdx.x;
  const int wave = tid >> 5, lane = tid & 31;
  const int lane16 = lane & 15, laneHi = lane >> 4;
  const int wm = wave & 1, wn = wave >> 1;   // 2 x 4 wave grid
  const int b = blockIdx.z, h = blockIdx.y;
  const int i0 = blockIdx.x * 32;

  const size_t bh = (size_t)b * 16 + h;
  const bf16_t* quG = qub + (bh * 512 + i0) * 64;
  const bf16_t* qvG = qvb + (bh * 512 + i0) * 64;
  const bf16_t* kG  = keyb + bh * 1024 * 64;
  const bf16_t* vG  = valT + bh * 64 * 1024;
  const bf16_t* rG  = rb + (size_t)h * 1024 * 64;

  { // stage qu/qv (2048 elems each, 8 per thread)
    const int idx = tid * 8; const int row = idx >> 6, col = idx & 63;
    *(uint4*)&s_qu[row][col] = *(const uint4*)&quG[row * 64 + col];
    *(uint4*)&s_qv[row][col] = *(const uint4*)&qvG[row * 64 + col];
  }
  if (tid < 32) { s_m[tid] = -3.0e30f; s_l[tid] = 0.0f; s_c[tid] = 1.0f; }

  CFrag accO;
#pragma unroll
  for (int e = 0; e < 8; ++e) accO.f[e] = 0.0f;

  for (int f0 = 0; f0 < 1024; f0 += 64) {
    __syncthreads();
#ifdef USE_TDM
    if (wave == 0) {
      // D# group0: count=1, lds_addr, 57-bit global addr, type=2 ("image")
      const unsigned long long ga =
          (unsigned long long)(const void*)(kG + (size_t)f0 * 64);
      const unsigned ldsa = (unsigned)(unsigned long long)(const void*)&s_k[0][0];
      u32x4 g0; i32x8 g1; i32x4 g2; i32x4 g3;
      g0[0] = 1u;                 // count=1 valid descriptor
      g0[1] = ldsa;               // lds_addr
      g0[2] = (unsigned)(ga & 0xFFFFFFFFull);
      g0[3] = ((unsigned)(ga >> 32) & 0x01FFFFFFu) | 0x80000000u; // type=2
      // D# group1: data_size=2B, tensor 64 x 1024, tile 64 x 64, stride0=64
      g1[0] = (int)(1u << 16);     // data_size=1 -> 2 bytes
      g1[1] = (int)(64u << 16);    // tensor_dim0[15:0] = 64 (bits 63:48)
      g1[2] = (int)(1024u << 16);  // tensor_dim0 hi | tensor_dim1[15:0]=1024
      g1[3] = (int)(64u << 16);    // tensor_dim1 hi | tile_dim0 = 64
      g1[4] = (int)64u;            // tile_dim1 = 64, tile_dim2 = 0
      g1[5] = (int)64u;            // tensor_dim0_stride = 64
      g1[6] = 0; g1[7] = 0;        // dim1_stride unused (2D)
      g2[0] = 0; g2[1] = 0; g2[2] = 0; g2[3] = 0;
      g3[0] = 0; g3[1] = 0; g3[2] = 0; g3[3] = 0;
      i32x8 g4; g4[0]=0; g4[1]=0; g4[2]=0; g4[3]=0; g4[4]=0; g4[5]=0; g4[6]=0; g4[7]=0;
      __builtin_amdgcn_tensor_load_to_lds(g0, g1, g2, g3, g4, 0);
    }
#else
    { // stage K tile (row-major copy == B-layout [j][d])
      const int idx = tid * 16; const int j = idx >> 6, col = idx & 63;
      *(uint4*)&s_k[j][col]     = *(const uint4*)&kG[(size_t)(f0 + j) * 64 + col];
      *(uint4*)&s_k[j][col + 8] = *(const uint4*)&kG[(size_t)(f0 + j) * 64 + col + 8];
    }
#endif
    { // stage V^T tile: s_v[d][j] <- valT[d][f0+j]  (coalesced both sides)
      const int idx = tid * 16; const int d = idx >> 6, col = idx & 63;
      *(uint4*)&s_v[d][col]     = *(const uint4*)&vG[(size_t)d * 1024 + f0 + col];
      *(uint4*)&s_v[d][col + 8] = *(const uint4*)&vG[(size_t)d * 1024 + f0 + col + 8];
    }
    { // stage r band, rows clamped into [0, F-1]
      const int jb0 = f0 + 511 - i0 - 31;
#pragma unroll
      for (int c = 0; c < 3; ++c) {
        const int idx = tid * 8 + c * 2048;
        const int jj = idx >> 6, col = idx & 63;
        int jg = jb0 + jj; jg = (jg < 0) ? 0 : ((jg > 1023) ? 1023 : jg);
        *(uint4*)&s_r[jj][col] = *(const uint4*)&rG[(size_t)jg * 64 + col];
      }
    }
    if (f0 + 64 < 1024) {
      __builtin_prefetch(kG + (size_t)(f0 + 64) * 64, 0, 0);
      __builtin_prefetch(vG + (size_t)(tid >> 2) * 1024 + f0 + 64, 0, 0);
    }
#ifdef USE_TDM
    if (wave == 0) __builtin_amdgcn_s_wait_tensorcnt(0);
#endif
    __syncthreads();

    // position band GEMM: (q+v) @ r^T -> 32x96
#pragma unroll
    for (int rr = 0; rr < 2; ++rr) {
      const int nb = wn + rr * 4;
      if (nb < 6) {
        CFrag acc;
#pragma unroll
        for (int e = 0; e < 8; ++e) acc.f[e] = 0.0f;
#pragma unroll
        for (int ks = 0; ks < 2; ++ks) {
          AFrag afr, bfr;
          const int arow = wm * 16 + lane16;
#pragma unroll
          for (int p = 0; p < 8; ++p) {
            const int kk = ((p < 4) ? (2 * p) : (16 + 2 * (p - 4))) + laneHi * 8 + ks * 32;
            afr.u[p] = *(const unsigned*)&s_qv[arow][kk];
          }
          const int bcol = nb * 16 + lane16;
#pragma unroll
          for (int p = 0; p < 8; ++p)
            bfr.u[p] = *(const unsigned*)&s_r[bcol][((p < 4) ? (2 * p) : (16 + 2 * (p - 4))) + laneHi * 8 + ks * 32];
          acc.v = __builtin_amdgcn_wmma_f32_16x16x32_bf16(
              false, afr.v, false, bfr.v, (short)0, acc.v, false, false);
        }
#pragma unroll
        for (int e = 0; e < 8; ++e)
          s_band[wm * 16 + laneHi * 8 + e][nb * 16 + lane16] = f2bf(acc.f[e]);
      }
    }
    __syncthreads();

    // content GEMM + shear + mask + scale
    {
      CFrag acc;
#pragma unroll
      for (int e = 0; e < 8; ++e) acc.f[e] = 0.0f;
#pragma unroll
      for (int ks = 0; ks < 2; ++ks) {
        AFrag afr, bfr;
        const int arow = wm * 16 + lane16;
#pragma unroll
        for (int p = 0; p < 8; ++p) {
          const int kk = ((p < 4) ? (2 * p) : (16 + 2 * (p - 4))) + laneHi * 8 + ks * 32;
          afr.u[p] = *(const unsigned*)&s_qu[arow][kk];
        }
        const int bcol = wn * 16 + lane16;
#pragma unroll
        for (int p = 0; p < 8; ++p)
          bfr.u[p] = *(const unsigned*)&s_k[bcol][((p < 4) ? (2 * p) : (16 + 2 * (p - 4))) + laneHi * 8 + ks * 32];
        acc.v = __builtin_amdgcn_wmma_f32_16x16x32_bf16(
            false, afr.v, false, bfr.v, (short)0, acc.v, false, false);
      }
#pragma unroll
      for (int e = 0; e < 8; ++e) {
        const int il = wm * 16 + laneHi * 8 + e;
        const int jl = wn * 16 + lane16;
        float sc = (acc.f[e] + bf2f(s_band[il][jl + 31 - il])) * 0.125f;
        const int kcol = f0 + jl, qrow = i0 + il;
        if (kcol > qrow + 512) sc = -3.0e30f;   // causal TXL mask (k >= q+P+1)
        s_S[il][jl] = sc;
      }
    }
    __syncthreads();

    // online softmax (one thread per q-row)
    if (tid < 32) {
      const int r = tid;
      const float mold = s_m[r];
      float mt = mold;
      for (int j = 0; j < 64; ++j) mt = fmaxf(mt, s_S[r][j]);
      const float corr = __expf(mold - mt);
      float l = s_l[r] * corr;
      for (int j = 0; j < 64; ++j) {
        const float pv = __expf(s_S[r][j] - mt);
        l += pv;
        s_p[r][j] = f2bf(pv);
      }
      s_m[r] = mt; s_l[r] = l; s_c[r] = corr;
    }
    __syncthreads();

    // rescale O, then O += Pexp @ V
#pragma unroll
    for (int e = 0; e < 8; ++e) accO.f[e] *= s_c[wm * 16 + laneHi * 8 + e];
#pragma unroll
    for (int ks = 0; ks < 2; ++ks) {
      AFrag afr, bfr;
      const int arow = wm * 16 + lane16;
#pragma unroll
      for (int p = 0; p < 8; ++p) {
        const int kk = ((p < 4) ? (2 * p) : (16 + 2 * (p - 4))) + laneHi * 8 + ks * 32;
        afr.u[p] = *(const unsigned*)&s_p[arow][kk];
      }
      const int bcol = wn * 16 + lane16;   // dh column
#pragma unroll
      for (int p = 0; p < 8; ++p)
        bfr.u[p] = *(const unsigned*)&s_v[bcol][((p < 4) ? (2 * p) : (16 + 2 * (p - 4))) + laneHi * 8 + ks * 32];
      accO.v = __builtin_amdgcn_wmma_f32_16x16x32_bf16(
          false, afr.v, false, bfr.v, (short)0, accO.v, false, false);
    }
  }

  // epilogue: av[(q*B+b)*1024 + h*64 + d] = O / l
#pragma unroll
  for (int e = 0; e < 8; ++e) {
    const int il = wm * 16 + laneHi * 8 + e;
    const float invl = 1.0f / s_l[il];
    const int qrow = i0 + il;
    const int dcol = wn * 16 + lane16;
    av[((size_t)qrow * 8 + b) * 1024 + h * 64 + dcol] = accO.f[e] * invl;
  }
}

// ---------------------------------------------------------------------------
// Host orchestration
// ---------------------------------------------------------------------------
static inline int ew_grid(long long n) {
  long long g = (n + 255) / 256;
  if (g > 8192) g = 8192;
  return (int)g;
}

extern "C" void kernel_launch(void* const* d_in, const int* in_sizes, int n_in,
                              void* d_out, int out_size, void* d_ws, size_t ws_size,
                              hipStream_t stream) {
  (void)in_sizes; (void)n_in; (void)out_size; (void)ws_size;

  const float* inputs = (const float*)d_in[0];
  const float* pos    = (const float*)d_in[1];
  const float* u_p    = (const float*)d_in[2];
  const float* v_p    = (const float*)d_in[3];
  const float* memory = (const float*)d_in[4];
  // d_in[5] = mask (recomputed analytically on-device)
  const float* Wkv = (const float*)d_in[6];   const float* bkv = (const float*)d_in[7];
  const float* Wq  = (const float*)d_in[8];   const float* bq  = (const float*)d_in[9];
  const float* Wpos= (const float*)d_in[10];  const float* bpos= (const float*)d_in[11];
  const float* Wproj=(const float*)d_in[12];  const float* bproj=(const float*)d_in[13];
  const float* ln1g=(const float*)d_in[14];   const float* ln1b=(const float*)d_in[15];
  const float* ln2g=(const float*)d_in[16];   const float* ln2b=(const float*)d_in[17];
  const float* W1 = (const float*)d_in[18];   const float* b1  = (const float*)d_in[19];
  const float* W2 = (const float*)d_in[20];   const float* b2  = (const float*)d_in[21];
  const float* g1W[6]; for (int j = 0; j < 6; ++j) g1W[j] = (const float*)d_in[22 + j];
  const float* g1bgp = (const float*)d_in[28];
  const float* g2W[6]; for (int j = 0; j < 6; ++j) g2W[j] = (const float*)d_in[29 + j];
  const float* g2bgp = (const float*)d_in[35];

  const size_t D = 1024, HID = 4096, Q = 512, F = 1024, Bb = 8, Hh = 16, DH = 64;
  const size_t MQ = Q * Bb;   // 4096 rows
  const size_t MF = F * Bb;   // 8192 rows

  // --- workspace bump allocator (lifetime-based aliases noted below) ---
  char* base = (char*)d_ws;
  size_t off = 0;
  auto alloc = [&](size_t bytes) -> char* {
    size_t o = (off + 255) & ~(size_t)255;
    off = o + bytes;
    return base + o;
  };

  // bf16 weights, all stored TRANSPOSED [N][K]
  bf16_t* wkv_b  = (bf16_t*)alloc(D * 2048 * 2);
  bf16_t* wq_b   = (bf16_t*)alloc(D * 1024 * 2);
  bf16_t* wpos_b = (bf16_t*)alloc(D * 1024 * 2);
  bf16_t* wproj_b= (bf16_t*)alloc(1024 * 1024 * 2);
  bf16_t* w1_b   = (bf16_t*)alloc(D * HID * 2);
  bf16_t* w2_b   = (bf16_t*)alloc(HID * D * 2);
  bf16_t* g1w_b[6]; for (int j = 0; j < 6; ++j) g1w_b[j] = (bf16_t*)alloc(1024 * 1024 * 2);
  bf16_t* g2w_b[6]; for (int j = 0; j < 6; ++j) g2w_b[j] = (bf16_t*)alloc(1024 * 1024 * 2);
  bf16_t* posb = (bf16_t*)alloc(F * D * 2);

  bf16_t* x1b  = (bf16_t*)alloc(MF * D * 2);          // dead after kv GEMM
  bf16_t* inb  = (bf16_t*)alloc(MQ * D * 2);
  bf16_t* keyb = (bf16_t*)alloc(Bb * Hh * F * DH * 2); // keyb+valb contiguous:
  bf16_t* valb = (bf16_t*)alloc(Bb * Hh * F * DH * 2); //   reused as m1b (32MB)
  bf16_t* qub  = (bf16_t*)alloc(MQ * D * 2);           // reused as rxb
  bf16_t* qvb  = (bf16_t*)alloc(MQ * D * 2);           // reused as x2b
  bf16_t* rbuf = (bf16_t*)alloc(Hh * F * DH * 2);
  bf16_t* avb  = (bf16_t*)alloc(MQ * D * 2);

  float* kvf = (float*)alloc(MF * 2048 * 4);           // reused as m1f (64MB)
  float* qf  = (float*)alloc(MQ * D * 4);              // reused as m2f
  float* rf  = (float*)alloc(F * D * 4);
  float* av  = (float*)alloc(MQ * D * 4);              // reused as a1
  float* tr  = (float*)alloc(MQ * D * 4);
  float* tz  = (float*)alloc(MQ * D * 4);
  float* th  = (float*)alloc(MQ * D * 4);
  float* o1  = (float*)alloc(MQ * D * 4);

  // aliases
  bf16_t* yb  = x1b;               // bf16 of gate "y"
  bf16_t* o1b = x1b + MQ * D;      // bf16 of o1 (second half of x1b region)
  bf16_t* rxb = qub;
  bf16_t* x2b = qvb;
  bf16_t* m1b = keyb;              // spans keyb+valb
  float*  m1f = kvf;
  float*  m2f = qf;
  float*  a1  = av;
  float*  outp = (float*)d_out;

  auto conv = [&](const float* s, bf16_t* d, long long n) {
    convert_f32_bf16_kernel<<<ew_grid(n), 256, 0, stream>>>(s, d, (int)n);
  };
  auto tconv = [&](const float* s, bf16_t* d, int K, int N) {
    tconv_kernel<<<dim3((unsigned)(N / 32), (unsigned)(K / 32)), 256, 0, stream>>>(s, d, K, N);
  };
  auto gemm = [&](const bf16_t* A, const bf16_t* WT, const float* bias,
                  const float* Cin, float* Cout, int M, int N, int K, int act) {
    dim3 g((unsigned)(N / GBN), (unsigned)(M / GBM));
    gemm_bf16_kernel<<<g, 256, 0, stream>>>(A, WT, bias, Cin, Cout, M, N, K, act);
  };

  const long long nQD = (long long)(MQ * D);

  // --- weight conversions to transposed bf16; activations to row-major bf16 ---
  tconv(Wkv, wkv_b, 1024, 2048);
  tconv(Wq, wq_b, 1024, 1024);
  tconv(Wpos, wpos_b, 1024, 1024);
  tconv(Wproj, wproj_b, 1024, 1024);
  tconv(W1, w1_b, 1024, 4096);
  tconv(W2, w2_b, 4096, 1024);
  for (int j = 0; j < 6; ++j) tconv(g1W[j], g1w_b[j], 1024, 1024);
  for (int j = 0; j < 6; ++j) tconv(g2W[j], g2w_b[j], 1024, 1024);
  conv(pos, posb, (long long)(F * D));
  conv(inputs, inb, nQD);

  // --- LN1 over concat([memory, inputs]) -> x1 bf16 ---
  ln_rows_kernel<<<(unsigned)MF, 256, 0, stream>>>(memory, inputs, (int)(512 * Bb),
                                                   ln1g, ln1b, x1b);

  // --- K/V, Q, R projections ---
  gemm(x1b, wkv_b, bkv, nullptr, kvf, (int)MF, 2048, 1024, 0);
  kv_scatter_kernel<<<ew_grid(8LL * 16 * 1024 * 64), 256, 0, stream>>>(kvf, keyb, valb);

  gemm(inb, wq_b, bq, nullptr, qf, (int)MQ, 1024, 1024, 0);
  quqv_kernel<<<ew_grid(nQD), 256, 0, stream>>>(qf, u_p, v_p, qub, qvb);

  gemm(posb, wpos_b, bpos, nullptr, rf, 1024, 1024, 1024, 0);
  r_scatter_kernel<<<ew_grid(16LL * 1024 * 64), 256, 0, stream>>>(rf, rbuf);

  // --- fused attention ---
  attn_kernel<<<dim3(16, 16, 8), 256, 0, stream>>>(qub, qvb, keyb, valb, rbuf, av);

  // --- output projection + ReLU -> a1 = y for gate1 ---
  conv(av, avb, nQD);
  gemm(avb, wproj_b, bproj, nullptr, a1, (int)MQ, 1024, 1024, 1);
  conv(a1, yb, nQD);

  // --- GRU gate 1 (x = inputs, y = a1) ---
  gemm(yb, g1w_b[0], nullptr, nullptr, tr, (int)MQ, 1024, 1024, 0);  // y@Wr
  gemm(inb, g1w_b[1], nullptr, tr, tr, (int)MQ, 1024, 1024, 0);      // + x@Ur
  sigmoid_kernel<<<ew_grid(nQD), 256, 0, stream>>>(tr, nullptr, (int)nQD);
  gemm(yb, g1w_b[2], nullptr, nullptr, tz, (int)MQ, 1024, 1024, 0);  // y@Wz
  gemm(inb, g1w_b[3], nullptr, tz, tz, (int)MQ, 1024, 1024, 0);      // + x@Uz
  sigmoid_kernel<<<ew_grid(nQD), 256, 0, stream>>>(tz, g1bgp, (int)nQD);
  rx_build_kernel<<<ew_grid(nQD), 256, 0, stream>>>(tr, inputs, rxb, (int)nQD);
  gemm(yb, g1w_b[4], nullptr, nullptr, th, (int)MQ, 1024, 1024, 0);  // y@Wg
  gemm(rxb, g1w_b[5], nullptr, th, th, (int)MQ, 1024, 1024, 0);      // + (r*x)@Ug
  tanh_kernel<<<ew_grid(nQD), 256, 0, stream>>>(th, (int)nQD);
  gate_combine_kernel<<<ew_grid(nQD), 256, 0, stream>>>(inputs, tz, th, o1, (int)nQD);

  // --- LN2 + MLP ---
  ln_rows_kernel<<<(unsigned)MQ, 256, 0, stream>>>(o1, nullptr, (int)MQ, ln2g, ln2b, x2b);
  gemm(x2b, w1_b, b1, nullptr, m1f, (int)MQ, (int)HID, 1024, 1);
  conv(m1f, m1b, (long long)(MQ * HID));
  gemm(m1b, w2_b, b2, nullptr, m2f, (int)MQ, 1024, (int)HID, 1);

  // --- GRU gate 2 (x = o1, y = m2) -> d_out ---
  conv(m2f, yb, nQD);
  conv(o1, o1b, nQD);
  gemm(yb, g2w_b[0], nullptr, nullptr, tr, (int)MQ, 1024, 1024, 0);
  gemm(o1b, g2w_b[1], nullptr, tr, tr, (int)MQ, 1024, 1024, 0);
  sigmoid_kernel<<<ew_grid(nQD), 256, 0, stream>>>(tr, nullptr, (int)nQD);
  gemm(yb, g2w_b[2], nullptr, nullptr, tz, (int)MQ, 1024, 1024, 0);
  gemm(o1b, g2w_b[3], nullptr, tz, tz, (int)MQ, 1024, 1024, 0);
  sigmoid_kernel<<<ew_grid(nQD), 256, 0, stream>>>(tz, g2bgp, (int)nQD);
  rx_build_kernel<<<ew_grid(nQD), 256, 0, stream>>>(tr, o1, rxb, (int)nQD);
  gemm(yb, g2w_b[4], nullptr, nullptr, th, (int)MQ, 1024, 1024, 0);
  gemm(rxb, g2w_b[5], nullptr, th, th, (int)MQ, 1024, 1024, 0);
  tanh_kernel<<<ew_grid(nQD), 256, 0, stream>>>(th, (int)nQD);
  gate_combine_kernel<<<ew_grid(nQD), 256, 0, stream>>>(o1, tz, th, outp, (int)nQD);
}
